// AttentionLayer_70394513981647
// MI455X (gfx1250) — matrix operand order repaired and checked by reading, not compiled, not run on hardware
//
#include <hip/hip_runtime.h>
#include <math.h>

typedef __bf16 bf16;
typedef __attribute__((ext_vector_type(16))) __bf16 v16bf;
typedef __attribute__((ext_vector_type(8)))  __bf16 v8bf;
typedef __attribute__((ext_vector_type(8)))  float  v8f;

#define B_ 64
#define L_ 512
#define R_ 512
#define H_ 512
#define D_ 256

// ---------------------------------------------------------------------------
// K0: W (H x D, f32) -> Wt (D x H, bf16) so a WMMA B-fragment (one output
// column, 16 consecutive K values) is a single contiguous 32-byte load.
// ---------------------------------------------------------------------------
__global__ __launch_bounds__(256)
void wt_kernel(const float* __restrict__ W, bf16* __restrict__ Wt) {
    int idx = blockIdx.x * 256 + threadIdx.x;   // 0 .. H*D-1 exactly
    int h = idx / D_;
    int d = idx % D_;
    Wt[d * H_ + h] = (bf16)W[h * D_ + d];
}

// ===========================================================================
// K1: fused projection GEMM (lt and rt), double-buffered K pipeline.
// ===========================================================================
struct ProjFrag {
    float4 f0, f1, f2, f3;   // raw f32 A data (16 values, contiguous K: k = 16h + s)
    v16bf  bb[4];            // 4 B fragments
};

// arow already includes +kb; bcol already includes +ko.
__device__ __forceinline__
void proj_load(ProjFrag& fr, const float* __restrict__ arow,
               const bf16* __restrict__ bcol, int k0) {
    fr.f0 = *(const float4*)(arow + k0);
    fr.f1 = *(const float4*)(arow + k0 + 4);
    fr.f2 = *(const float4*)(arow + k0 + 8);
    fr.f3 = *(const float4*)(arow + k0 + 12);
#pragma unroll
    for (int t = 0; t < 4; ++t)
        fr.bb[t] = *(const v16bf*)(bcol + (long)t * 16 * H_ + k0);
}

__device__ __forceinline__
void proj_mma(const ProjFrag& fr, v8f acc[4]) {
    float af[16] = {fr.f0.x, fr.f0.y, fr.f0.z, fr.f0.w,
                    fr.f1.x, fr.f1.y, fr.f1.z, fr.f1.w,
                    fr.f2.x, fr.f2.y, fr.f2.z, fr.f2.w,
                    fr.f3.x, fr.f3.y, fr.f3.z, fr.f3.w};
    v16bf a;
#pragma unroll
    for (int i = 0; i < 16; ++i) a[i] = (bf16)af[i];
#pragma unroll
    for (int t = 0; t < 4; ++t)
        acc[t] = __builtin_amdgcn_wmma_f32_16x16x32_bf16(
            false, a, false, fr.bb[t], (short)0, acc[t], false, false);
}

__global__ __launch_bounds__(256)
void proj_kernel(const float* __restrict__ lt, const float* __restrict__ rt,
                 const bf16* __restrict__ Wt, const float* __restrict__ diag,
                 bf16* __restrict__ out_lt, bf16* __restrict__ out_rt) {
    const int tid    = threadIdx.x;
    const int lane   = tid & 31;
    const int wave   = tid >> 5;
    const int rowgrp = wave >> 2;          // 0..1
    const int nbase  = (wave & 3) * 64;    // 0,64,128,192

    const long rowbase = (long)blockIdx.x * 32 + rowgrp * 16;
    const bool  isLt   = rowbase < (long)B_ * L_;
    const float* src   = isLt ? lt : rt;
    bf16*        dst   = isLt ? out_lt : out_rt;
    const long  srcrow = isLt ? rowbase : rowbase - (long)B_ * L_;

    const int m16 = lane & 15;
    const int kb  = (lane >> 4) * 16;   // A layout: halves = K{kb..kb+15}
    const int ko  = (lane >> 4) * 16;   // B layout: halves = K{ko..ko+15}

    const float* arow = src + (srcrow + m16) * (long)H_ + kb;
    const bf16*  bcol = Wt + (long)(nbase + m16) * H_ + ko;

    v8f acc[4] = {};

    ProjFrag fA, fB;
    proj_load(fA, arow, bcol, 0);
#pragma unroll 1
    for (int k0 = 0; k0 < H_; k0 += 64) {
        proj_load(fB, arow, bcol, k0 + 32);
        proj_mma(fA, acc);
        if (k0 + 64 < H_) proj_load(fA, arow, bcol, k0 + 64);
        proj_mma(fB, acc);
    }

    // ---- epilogue: tanh (+ diag), store bf16 ----
    const int mofs = (lane < 16) ? 0 : 8;
#pragma unroll
    for (int t = 0; t < 4; ++t) {
        int n = nbase + t * 16 + m16;
        float dscale = isLt ? diag[n] : 1.0f;
#pragma unroll
        for (int v = 0; v < 8; ++v) {
            long row = srcrow + v + mofs;
            float val = tanhf(acc[t][v]) * dscale;
            dst[row * D_ + n] = (bf16)val;
        }
    }
}

// ===========================================================================
// K2: per-batch S = a_lt (L x D) @ a_rt^T, fused row-softmax.
// Double-buffered K pipeline; 16x512 f32 LDS tile; b128 output stores.
// ===========================================================================
struct AttnFrag {
    v8bf  alo, ahi;
    v16bf bb[4];
};

// arow already includes +kb; rrow already includes +ko.
__device__ __forceinline__
void attn_load(AttnFrag& fr, const bf16* __restrict__ arow,
               const bf16* __restrict__ rrow, int k0) {
    fr.alo = *(const v8bf*)(arow + k0);
    fr.ahi = *(const v8bf*)(arow + k0 + 8);
#pragma unroll
    for (int t = 0; t < 4; ++t)
        fr.bb[t] = *(const v16bf*)(rrow + (long)t * 16 * D_ + k0);
}

__device__ __forceinline__
void attn_mma(const AttnFrag& fr, v8f acc[4]) {
    v16bf a;
#pragma unroll
    for (int i = 0; i < 8; ++i) { a[i] = fr.alo[i]; a[8 + i] = fr.ahi[i]; }
#pragma unroll
    for (int t = 0; t < 4; ++t)
        acc[t] = __builtin_amdgcn_wmma_f32_16x16x32_bf16(
            false, a, false, fr.bb[t], (short)0, acc[t], false, false);
}

__global__ __launch_bounds__(256)
void attn_kernel(const bf16* __restrict__ a_lt, const bf16* __restrict__ a_rt,
                 float* __restrict__ out) {
    __shared__ float s_tile[16 * 512];
    __shared__ float s_max[16 * 16];
    __shared__ float s_sum[16 * 16];

    const int tid  = threadIdx.x;
    const int lane = tid & 31;
    const int wave = tid >> 5;

    const int  b    = blockIdx.x >> 5;        // batch 0..63
    const int  lgrp = blockIdx.x & 31;        // L row-group 0..31
    const long l0   = (long)b * L_ + lgrp * 16;

    const int m16   = lane & 15;
    const int kb    = (lane >> 4) * 16;
    const int ko    = (lane >> 4) * 16;
    const int nbase = wave * 64;

    const bf16* arow = a_lt + (l0 + m16) * (long)D_ + kb;
    const bf16* rrow = a_rt + ((long)b * R_ + nbase + m16) * D_ + ko;

    v8f acc[4] = {};

    AttnFrag fA, fB;
    attn_load(fA, arow, rrow, 0);
#pragma unroll 1
    for (int k0 = 0; k0 < D_; k0 += 64) {
        attn_load(fB, arow, rrow, k0 + 32);
        attn_mma(fA, acc);
        if (k0 + 64 < D_) attn_load(fA, arow, rrow, k0 + 64);
        attn_mma(fB, acc);
    }

    // ---- dump accumulators to LDS (16 x 512 f32) ----
    const int mofs = (lane < 16) ? 0 : 8;
#pragma unroll
    for (int t = 0; t < 4; ++t) {
        int col = nbase + t * 16 + m16;
#pragma unroll
        for (int v = 0; v < 8; ++v)
            s_tile[(v + mofs) * 512 + col] = acc[t][v];
    }
    __syncthreads();

    // ---- softmax: 16 rows x 16 threads, 32 contiguous cols per thread ----
    const int row = tid >> 4;     // 0..15
    const int sub = tid & 15;     // 0..15
    float* rowp = &s_tile[row * 512 + sub * 32];

    float4 v4[8];
#pragma unroll
    for (int i = 0; i < 8; ++i) v4[i] = *(const float4*)(rowp + i * 4);

    float pmax = v4[0].x;
#pragma unroll
    for (int i = 0; i < 8; ++i)
        pmax = fmaxf(pmax, fmaxf(fmaxf(v4[i].x, v4[i].y), fmaxf(v4[i].z, v4[i].w)));
    s_max[row * 16 + sub] = pmax;
    __syncthreads();

    float rmax = s_max[row * 16 + 0];
#pragma unroll
    for (int i = 1; i < 16; ++i) rmax = fmaxf(rmax, s_max[row * 16 + i]);

    float psum = 0.0f;
#pragma unroll
    for (int i = 0; i < 8; ++i) {
        v4[i].x = __expf(v4[i].x - rmax);
        v4[i].y = __expf(v4[i].y - rmax);
        v4[i].z = __expf(v4[i].z - rmax);
        v4[i].w = __expf(v4[i].w - rmax);
        psum += v4[i].x + v4[i].y + v4[i].z + v4[i].w;
    }
    s_sum[row * 16 + sub] = psum;
    __syncthreads();

    float rsum = 0.0f;
#pragma unroll
    for (int i = 0; i < 16; ++i) rsum += s_sum[row * 16 + i];
    const float inv = 1.0f / rsum;

    float* orow = out + (l0 + row) * (long)R_ + sub * 32;
#pragma unroll
    for (int i = 0; i < 8; ++i) {
        float4 o = {v4[i].x * inv, v4[i].y * inv, v4[i].z * inv, v4[i].w * inv};
        *(float4*)(orow + i * 4) = o;
    }
}

// ---------------------------------------------------------------------------
extern "C" void kernel_launch(void* const* d_in, const int* in_sizes, int n_in,
                              void* d_out, int out_size, void* d_ws, size_t ws_size,
                              hipStream_t stream) {
    const float* lt   = (const float*)d_in[0];   // (B, L, H) f32
    const float* rt   = (const float*)d_in[1];   // (B, R, H) f32
    const float* W    = (const float*)d_in[2];   // (H, D)    f32
    const float* diag = (const float*)d_in[3];   // (1,1,D)   f32
    float* out = (float*)d_out;                  // (B, L, R) f32

    char* ws = (char*)d_ws;
    bf16* Wt  = (bf16*)ws;                                         // 256 KB
    bf16* alt = (bf16*)(ws + 512 * 1024);                          // 16 MB
    bf16* art = (bf16*)(ws + 512 * 1024 + (size_t)B_ * L_ * D_ * 2);

    wt_kernel  <<<(H_ * D_) / 256, 256, 0, stream>>>(W, Wt);
    proj_kernel<<<(2 * B_ * L_) / 32, 256, 0, stream>>>(lt, rt, Wt, diag, alt, art);
    attn_kernel<<<B_ * 32, 256, 0, stream>>>(alt, art, out);
}
